// LowRankMultiheadAttention_75651553952401
// MI455X (gfx1250) — matrix operand order repaired and checked by reading, not compiled, not run on hardware
//
#include <hip/hip_runtime.h>
#include <hip/hip_bf16.h>
#include <stdint.h>

typedef _Float16 v16h __attribute__((ext_vector_type(16)));
typedef float    v8f  __attribute__((ext_vector_type(8)));
typedef _Float16 h2   __attribute__((ext_vector_type(2)));

#define IN_DIM 1024
#define EMBED  1024
#define HEADS  16
#define HDIM   64
#define BATCH  32
#define SEQ    400
#define NROWS  (BATCH * SEQ)   // 12800
#define LDS_STRIDE 40          // 32 data halfs + 8 pad (keeps 16B align, spreads banks)

// ---- WMMA fragment index helpers (CDNA5 ISA 7.12.2, wave32) ----
// A (16x32 f16): lane lo = row M; pair v -> K = hi*16 + 2v (this code's order, not the ISA table)
static __device__ __forceinline__ int a_k0(int v, int hi) {
  return (hi << 4) + (v << 1);
}
// B (32x16 f16): lane lo = col N; pair v -> K = hi*16 + 2v (this code's order, not the ISA table)
static __device__ __forceinline__ int b_k0(int v, int hi) {
  return (hi << 4) + (v << 1);
}
static __device__ __forceinline__ v8f wmma_f16(v16h a, v16h b, v8f c) {
  return __builtin_amdgcn_wmma_f32_16x16x32_f16(false, a, false, b, (short)0, c,
                                                false, false);
}
// async copy: 16B per lane, global -> LDS (tracked by ASYNCcnt)
static __device__ __forceinline__ void async_b128(uint32_t lds_off,
                                                  const _Float16* g) {
  asm volatile("global_load_async_to_lds_b128 %0, %1, off"
               :: "v"(lds_off), "v"(g) : "memory");
}
// DPP16 max-butterfly step: fmax with lane from CTRL permutation (VALU only)
template <int CTRL>
static __device__ __forceinline__ float maxdpp(float x) {
  int y = __builtin_amdgcn_update_dpp(0, __float_as_int(x), CTRL, 0xf, 0xf, true);
  return fmaxf(x, __int_as_float(y));
}
// full 16-lane row max: xor1, xor2, then ^7 (half mirror), ^15 (mirror)
static __device__ __forceinline__ float rowmax16(float x) {
  x = maxdpp<0xB1>(x);    // quad_perm [1,0,3,2]
  x = maxdpp<0x4E>(x);    // quad_perm [2,3,0,1]
  x = maxdpp<0x141>(x);   // ROW_HALF_MIRROR (lane ^ 7)
  x = maxdpp<0x140>(x);   // ROW_MIRROR      (lane ^ 15)
  return x;
}

// ------------------- stage 1: f32 -> f16 conversion -------------------
__global__ void cvt_f32_f16(const float* __restrict__ src,
                            _Float16* __restrict__ dst, int n) {
  int i = blockIdx.x * blockDim.x + threadIdx.x;
  if (i < n) dst[i] = (_Float16)src[i];
}

// ------------- stage 2: pos[s,e] = pe[s,:].Wp[e,:] + bp[e] -------------
__global__ void pos_proj(const float* __restrict__ pe, const float* __restrict__ Wp,
                         const float* __restrict__ bp, float* __restrict__ pos) {
  int e = blockIdx.x * blockDim.x + threadIdx.x;   // grid.x = EMBED/256
  int s = blockIdx.y;                              // grid.y = SEQ
  const float* per = pe + (size_t)s * IN_DIM;
  const float* wr  = Wp + (size_t)e * IN_DIM;
  float acc = 0.f;
  for (int i = 0; i < IN_DIM; ++i) acc = fmaf(per[i], wr[i], acc);
  pos[(size_t)s * EMBED + e] = acc + bp[e];
}

// ------ stage 3/5: WMMA GEMM  Out[n,e] = A[n,:].W[e,:] + epilogue ------
// W tile (128 cols x 32 k) double-buffered in LDS via async copies, shared by
// the 4 waves of the block. MODE 0: f16 out + bias + pos (QKV projection);
// MODE 1: f32 out, (acc+bias)*gate (output projection).
template <bool AF16, int MODE>
__global__ void __launch_bounds__(128)
gemm_wmma(const void* __restrict__ Ain, const _Float16* __restrict__ W,
          const float* __restrict__ bias, const float* __restrict__ pos,
          const float* __restrict__ gate, void* __restrict__ Out) {
  __shared__ _Float16 Wt[2][128][LDS_STRIDE];

  const int lane  = threadIdx.x;                    // blockDim.x == 32
  const int hi    = lane >> 4;
  const int lo    = lane & 15;
  const int tid   = threadIdx.y * 32 + lane;        // 0..127
  const int nt    = blockIdx.y * 4 + threadIdx.y;   // n-tile 0..799
  const int eBase = blockIdx.x * 128;
  const int nrow  = nt * 16;

  // this thread stages output-column eBase+tid: 32 halfs = 4 x b128 per k-step
  const _Float16* gW = W + (size_t)(eBase + tid) * IN_DIM;

  {  // prime buffer 0
    uint32_t l0 = (uint32_t)(uintptr_t)&Wt[0][tid][0];
#pragma unroll
    for (int j = 0; j < 4; ++j) async_b128(l0 + j * 16, gW + j * 8);
  }

  v8f acc[8] = {};
  const size_t arow = (size_t)(nrow + lo) * IN_DIM;

  for (int ks = 0; ks < IN_DIM / 32; ++ks) {
    const int sel = ks & 1;
    if (ks + 1 < IN_DIM / 32) {  // issue next tile, then wait for current
      uint32_t l1 = (uint32_t)(uintptr_t)&Wt[sel ^ 1][tid][0];
      const _Float16* gn = gW + (ks + 1) * 32;
#pragma unroll
      for (int j = 0; j < 4; ++j) async_b128(l1 + j * 16, gn + j * 8);
      asm volatile("s_wait_asynccnt 4" ::: "memory");
    } else {
      asm volatile("s_wait_asynccnt 0" ::: "memory");
    }
    __syncthreads();  // current buffer visible to all 4 waves

    const int kk = ks * 32;
    v16h a;
    if (AF16) {
      const _Float16* Af = (const _Float16*)Ain + arow + kk;
#pragma unroll
      for (int v = 0; v < 8; ++v) {
        h2 p = *(const h2*)(Af + a_k0(v, hi));
        a[2 * v] = p.x; a[2 * v + 1] = p.y;
      }
    } else {
      const float* Af = (const float*)Ain + arow + kk;
#pragma unroll
      for (int v = 0; v < 8; ++v) {
        int k0 = a_k0(v, hi);
        a[2 * v]     = (_Float16)Af[k0];
        a[2 * v + 1] = (_Float16)Af[k0 + 1];
      }
    }
#pragma unroll
    for (int t = 0; t < 8; ++t) {
      const _Float16* Wl = &Wt[sel][t * 16 + lo][0];
      v16h b;
#pragma unroll
      for (int v = 0; v < 8; ++v) {
        h2 p = *(const h2*)(Wl + b_k0(v, hi));
        b[2 * v] = p.x; b[2 * v + 1] = p.y;
      }
      acc[t] = wmma_f16(a, b, acc[t]);
    }
    __syncthreads();  // all reads done before buffer is overwritten
  }

#pragma unroll
  for (int t = 0; t < 8; ++t) {
    int e = eBase + t * 16 + lo;
#pragma unroll
    for (int r = 0; r < 8; ++r) {
      int n = nrow + r + 8 * hi;
      float val = acc[t][r] + bias[e];
      if (MODE == 0) {
        int s = n % SEQ;
        val += pos[(size_t)s * EMBED + e];
        ((_Float16*)Out)[(size_t)n * EMBED + e] = (_Float16)val;
      } else {
        ((float*)Out)[(size_t)n * EMBED + e] = val * gate[0];
      }
    }
  }
}

// --------------- stage 4: flash attention per (b, h) ---------------
// One key-tile step (32 keys). TAIL=true: keys [384,400) -> s1 tile fully OOB,
// s0 K-loads always in range, V rows OOB exactly when hi==1.
// o[0..3]: output d-subtiles; o[4]: row-sum tile (B = ones), so the sum
// reduction rides the tensor core; only the row max uses DPP.
template <bool TAIL>
static __device__ __forceinline__ void attn_kt(
    const int kbase, const int qbase,
    const _Float16* __restrict__ Kb, const _Float16* __restrict__ Vb,
    const float* __restrict__ mask, _Float16* __restrict__ P,
    const v16h aq0, const v16h aq1, const v16h bones,
    float* __restrict__ m, v8f* __restrict__ o, const int hi, const int lo) {
  v8f s0 = {}, s1 = {};
  {
    const _Float16* kr0 = Kb + (size_t)(kbase + lo) * EMBED;
    v16h bk;
#pragma unroll
    for (int v = 0; v < 8; ++v) {
      h2 p = *(const h2*)(kr0 + b_k0(v, hi));
      bk[2 * v] = p.x; bk[2 * v + 1] = p.y;
    }
    s0 = wmma_f16(aq0, bk, s0);
#pragma unroll
    for (int v = 0; v < 8; ++v) {
      h2 p = *(const h2*)(kr0 + 32 + b_k0(v, hi));
      bk[2 * v] = p.x; bk[2 * v + 1] = p.y;
    }
    s0 = wmma_f16(aq1, bk, s0);
    if (!TAIL) {
      const _Float16* kr1 = kr0 + (size_t)16 * EMBED;
#pragma unroll
      for (int v = 0; v < 8; ++v) {
        h2 p = *(const h2*)(kr1 + b_k0(v, hi));
        bk[2 * v] = p.x; bk[2 * v + 1] = p.y;
      }
      s1 = wmma_f16(aq0, bk, s1);
#pragma unroll
      for (int v = 0; v < 8; ++v) {
        h2 p = *(const h2*)(kr1 + 32 + b_k0(v, hi));
        bk[2 * v] = p.x; bk[2 * v + 1] = p.y;
      }
      s1 = wmma_f16(aq1, bk, s1);
    }
  }

  const int kcol0 = kbase + lo;
  const int kcol1 = kbase + 16 + lo;
#pragma unroll
  for (int r = 0; r < 8; ++r) {
    const int qrow = qbase + r + 8 * hi;
    const float* mrow = mask + (size_t)qrow * SEQ;
    float sc0 = s0[r] + mrow[kcol0];
    float sc1 = TAIL ? -1e30f : (s1[r] + mrow[kcol1]);
    const float rm = rowmax16(fmaxf(sc0, sc1));   // DPP butterfly, no LDS
    const float mn = fmaxf(m[r], rm);
    const float sc = __expf(m[r] - mn);
    m[r] = mn;
    const float p0 = __expf(sc0 - mn);
    const float p1 = TAIL ? 0.f : __expf(sc1 - mn);
    const int prow = r + 8 * hi;
    P[prow * 32 + lo]      = (_Float16)p0;
    P[prow * 32 + 16 + lo] = (_Float16)p1;
#pragma unroll
    for (int t = 0; t < 5; ++t) o[t][r] *= sc;
  }
  __asm__ volatile("s_wait_dscnt 0" ::: "memory");

  v16h pa;  // reload P as an A-fragment (16x32)
#pragma unroll
  for (int v = 0; v < 8; ++v) {
    h2 p = *(const h2*)(P + lo * 32 + a_k0(v, hi));
    pa[2 * v] = p.x; pa[2 * v + 1] = p.y;
  }
#pragma unroll
  for (int dt = 0; dt < 4; ++dt) {
    v16h bv;
#pragma unroll
    for (int v = 0; v < 8; ++v) {
      const int k0 = b_k0(v, hi);
      const size_t r0 = (size_t)(kbase + k0) * EMBED + dt * 16 + lo;
      _Float16 e0, e1;
      if (TAIL) {
        e0 = hi ? (_Float16)0.f : Vb[r0];
        e1 = hi ? (_Float16)0.f : Vb[r0 + EMBED];
      } else {
        e0 = Vb[r0];
        e1 = Vb[r0 + EMBED];
      }
      bv[2 * v] = e0; bv[2 * v + 1] = e1;
    }
    o[dt] = wmma_f16(pa, bv, o[dt]);
  }
  o[4] = wmma_f16(pa, bones, o[4]);   // row sums ride the tensor core
}

__global__ void __launch_bounds__(256)
attn_kernel(const _Float16* __restrict__ Q, const _Float16* __restrict__ K,
            const _Float16* __restrict__ V, const float* __restrict__ mask,
            _Float16* __restrict__ O) {
  __shared__ _Float16 Pslab[8][16 * 32];   // per-wave P tile (C->A relayout)

  const int bh   = blockIdx.x;             // BATCH*HEADS blocks
  const int b    = bh / HEADS;
  const int h    = bh % HEADS;
  const int tid  = threadIdx.x;
  const int wid  = tid >> 5;
  const int lane = tid & 31;
  const int hi   = lane >> 4;
  const int lo   = lane & 15;
  const size_t base = (size_t)b * SEQ * EMBED + (size_t)h * HDIM;
  const _Float16* Qb = Q + base;
  const _Float16* Kb = K + base;
  const _Float16* Vb = V + base;

  _Float16* P = Pslab[wid];
  v16h bones;
#pragma unroll
  for (int i = 0; i < 16; ++i) bones[i] = (_Float16)1.f;
  const _Float16 qscale = (_Float16)0.125f;   // 1/sqrt(64), exact in f16

  for (int qt = wid; qt < SEQ / 16; qt += 8) {
    const int qbase = qt * 16;

    v16h aq0, aq1;  // Q A-fragments d=[0,32), [32,64), pre-scaled
    {
      const _Float16* qr = Qb + (size_t)(qbase + lo) * EMBED;
#pragma unroll
      for (int v = 0; v < 8; ++v) {
        int k0 = a_k0(v, hi);
        h2 p0 = *(const h2*)(qr + k0);
        h2 p1 = *(const h2*)(qr + 32 + k0);
        aq0[2 * v] = p0.x * qscale; aq0[2 * v + 1] = p0.y * qscale;
        aq1[2 * v] = p1.x * qscale; aq1[2 * v + 1] = p1.y * qscale;
      }
    }

    float m[8];
    v8f o[5] = {};
#pragma unroll
    for (int r = 0; r < 8; ++r) m[r] = -1e30f;

    for (int kt = 0; kt < 12; ++kt)   // guard-free main loop
      attn_kt<false>(kt * 32, qbase, Kb, Vb, mask, P, aq0, aq1, bones, m, o, hi, lo);
    attn_kt<true>(384, qbase, Kb, Vb, mask, P, aq0, aq1, bones, m, o, hi, lo);

#pragma unroll
    for (int r = 0; r < 8; ++r) {
      const float inv = __builtin_amdgcn_rcpf(o[4][r]);  // v_rcp_f32, no IEEE div blob
      const int srow = qbase + r + 8 * hi;
#pragma unroll
      for (int dt = 0; dt < 4; ++dt)
        O[base + (size_t)srow * EMBED + dt * 16 + lo] = (_Float16)(o[dt][r] * inv);
    }
  }
}

// ------------------------------ launch ------------------------------
extern "C" void kernel_launch(void* const* d_in, const int* in_sizes, int n_in,
                              void* d_out, int out_size, void* d_ws, size_t ws_size,
                              hipStream_t stream) {
  const float* query = (const float*)d_in[0];
  const float* key   = (const float*)d_in[1];
  const float* value = (const float*)d_in[2];
  const float* amask = (const float*)d_in[3];
  const float* pe    = (const float*)d_in[4];
  const float* Wq = (const float*)d_in[5];  const float* bq = (const float*)d_in[6];
  const float* Wk = (const float*)d_in[7];  const float* bk = (const float*)d_in[8];
  const float* Wv = (const float*)d_in[9];  const float* bv = (const float*)d_in[10];
  const float* Wp = (const float*)d_in[11]; const float* bp = (const float*)d_in[12];
  const float* Wo = (const float*)d_in[13]; const float* bo = (const float*)d_in[14];
  const float* gates = (const float*)d_in[15];

  char* ws = (char*)d_ws;
  const size_t WSZ = (size_t)EMBED * IN_DIM * sizeof(_Float16);  // 2 MiB
  const size_t QSZ = (size_t)NROWS * EMBED * sizeof(_Float16);   // 25 MiB
  _Float16* Wq16 = (_Float16*)(ws);
  _Float16* Wk16 = (_Float16*)(ws + WSZ);
  _Float16* Wv16 = (_Float16*)(ws + 2 * WSZ);
  _Float16* Wo16 = (_Float16*)(ws + 3 * WSZ);
  float*    posf = (float*)(ws + 4 * WSZ);                       // 1.6 MB
  char* p0 = ws + 4 * WSZ +
             (((size_t)SEQ * EMBED * sizeof(float) + 255) & ~(size_t)255);
  _Float16* Q16 = (_Float16*)(p0);
  _Float16* K16 = (_Float16*)(p0 + QSZ);
  _Float16* V16 = (_Float16*)(p0 + 2 * QSZ);
  _Float16* O16 = (_Float16*)(p0 + 3 * QSZ);

  const int wn = EMBED * IN_DIM;
  cvt_f32_f16<<<(wn + 255) / 256, 256, 0, stream>>>(Wq, Wq16, wn);
  cvt_f32_f16<<<(wn + 255) / 256, 256, 0, stream>>>(Wk, Wk16, wn);
  cvt_f32_f16<<<(wn + 255) / 256, 256, 0, stream>>>(Wv, Wv16, wn);
  cvt_f32_f16<<<(wn + 255) / 256, 256, 0, stream>>>(Wo, Wo16, wn);

  pos_proj<<<dim3(EMBED / 256, SEQ), 256, 0, stream>>>(pe, Wp, bp, posf);

  dim3 gblock(32, 4);
  dim3 ggrid(EMBED / 128, NROWS / 16 / 4);   // (8, 200)
  gemm_wmma<false, 0><<<ggrid, gblock, 0, stream>>>(query, Wq16, bq, posf, gates, Q16);
  gemm_wmma<false, 0><<<ggrid, gblock, 0, stream>>>(key,   Wk16, bk, posf, gates, K16);
  gemm_wmma<false, 0><<<ggrid, gblock, 0, stream>>>(value, Wv16, bv, posf, gates, V16);

  attn_kernel<<<BATCH * HEADS, 256, 0, stream>>>(Q16, K16, V16, amask, O16);

  gemm_wmma<true, 1><<<ggrid, gblock, 0, stream>>>(O16, Wo16, bo, posf, gates, (float*)d_out);
}
